// YingGemAttention_29265907155205
// MI455X (gfx1250) — compile-verified, matrix operand order checked by reading, not run on hardware
//
#include <hip/hip_runtime.h>

// ---------------------------------------------------------------------------
// Sliding-window attention (RoPE, WIN=512, non-causal) for MI455X / gfx1250.
// All matmuls run on v_wmma_f32_16x16x32_bf16 (wave32 WMMA, f32 accumulate).
// All LDS fragment reads are vectorized ds_load_b128 (transposed staging for
// column-accessed operands). K-tile staging uses GLOBAL_LOAD_ASYNC_TO_LDS_B128
// when the toolchain exposes the builtin.
// ---------------------------------------------------------------------------

#define B_   2
#define T_   2048
#define H_   16
#define D_   64
#define C_   1024
#define M_   (B_ * T_)     // 4096 rows for every GEMM
#define WIN_ 512
#define NEGF (-1.0e30f)

typedef __attribute__((ext_vector_type(16))) __bf16 v16bf;
typedef __attribute__((ext_vector_type(8)))  __bf16 v8bf;
typedef __attribute__((ext_vector_type(8)))  float  v8f;
typedef int v4i_ __attribute__((vector_size(16)));   // builtin's pointee type

#if defined(__has_builtin)
#if __has_builtin(__builtin_amdgcn_global_load_async_to_lds_b128) && \
    __has_builtin(__builtin_amdgcn_s_wait_asynccnt)
#define USE_ASYNC_COPY 1
#endif
#endif

static __device__ __forceinline__ __bf16 f2bf(float f) {
  union { float f; unsigned u; } v; v.f = f;
  unsigned r = v.u + 0x7FFFu + ((v.u >> 16) & 1u);   // round-to-nearest-even
  unsigned short h = (unsigned short)(r >> 16);
  __bf16 o; __builtin_memcpy(&o, &h, 2); return o;
}
static __device__ __forceinline__ float bf2f(__bf16 b) {
  unsigned short h; __builtin_memcpy(&h, &b, 2);
  union { unsigned u; float f; } v; v.u = ((unsigned)h) << 16; return v.f;
}
static __device__ __forceinline__ v8f zero8() {
  v8f z = {0.f, 0.f, 0.f, 0.f, 0.f, 0.f, 0.f, 0.f}; return z;
}
static __device__ __forceinline__ v8bf ld8(const __bf16* p) {
  return *(const v8bf*)p;                     // 16B aligned by construction
}
static __device__ __forceinline__ v16bf cat8(v8bf lo, v8bf hi) {
  return __builtin_shufflevector(lo, hi, 0, 1, 2, 3, 4, 5, 6, 7,
                                 8, 9, 10, 11, 12, 13, 14, 15);
}
// packed 4x bf16 store to LDS (8B aligned by construction)
static __device__ __forceinline__ void st4(__bf16* dst, __bf16 a, __bf16 b,
                                           __bf16 c, __bf16 d) {
  union { unsigned short u[4]; uint2 v; } t;
  __builtin_memcpy(&t.u[0], &a, 2);
  __builtin_memcpy(&t.u[1], &b, 2);
  __builtin_memcpy(&t.u[2], &c, 2);
  __builtin_memcpy(&t.u[3], &d, 2);
  *(uint2*)dst = t.v;
}
#if USE_ASYNC_COPY
// per-lane async DMA, 16B global -> LDS (tracked by ASYNCcnt).
// Builtin expects (int4 AS1*, int4 AS3*, imm offset, imm cpol); integer
// round-trips: flat->LDS truncation matches LLVM's addrspacecast lowering.
static __device__ __forceinline__ void async_cp16(const __bf16* g, __bf16* l) {
  __builtin_amdgcn_global_load_async_to_lds_b128(
      (__attribute__((address_space(1))) v4i_*)(uintptr_t)g,
      (__attribute__((address_space(3))) v4i_*)(uintptr_t)l, 0, 0);
}
#endif

// ---------------------------------------------------------------------------
// GEMM: [4096 x 1024] @ [1024 x 1024] + bias.
// MODE 0: A is fp32 (x), output bf16 in head-major [B,H,T,D]  (Q/K/V proj)
// MODE 1: A is bf16 (attn out), output fp32 row-major          (O proj)
// Block tile 128x128, 8 waves, per-wave 32x64, K-step 32.
// W tile stored transposed in LDS so both operand reads are ds_load_b128.
// ---------------------------------------------------------------------------
template <int MODE>
__global__ __launch_bounds__(256)
void gemm_wmma(const void* __restrict__ Ap, const float* __restrict__ W,
               const float* __restrict__ bias, void* __restrict__ outp)
{
  __shared__ __attribute__((aligned(16))) __bf16 As [128 * 48];  // [m][k]+pad
  __shared__ __attribute__((aligned(16))) __bf16 BsT[128 * 48];  // [n][k]+pad

  const int bm   = blockIdx.y * 128;
  const int bn   = blockIdx.x * 128;
  const int tid  = threadIdx.x;
  const int lane = tid & 31;
  const int w    = tid >> 5;
  const int wm   = (w & 3) * 32;
  const int wn   = (w >> 2) * 64;
  const int half = lane >> 4;
  const int l16  = lane & 15;

  v8f acc[2][4];
  for (int i = 0; i < 2; ++i)
    for (int j = 0; j < 4; ++j) acc[i][j] = zero8();

  for (int k0 = 0; k0 < C_; k0 += 32) {
    // prefetch next W tile (global_prefetch path)
    if (k0 + 32 < C_)
      __builtin_prefetch((const void*)(W + (size_t)(k0 + 32 + (tid & 31)) * C_ +
                                       bn + (tid >> 5) * 16), 0, 1);
    __syncthreads();
    // ---- stage A tile into LDS as bf16 (row-major in k) ----
    if constexpr (MODE == 0) {
      const float* A = (const float*)Ap;
#pragma unroll
      for (int p = 0; p < 4; ++p) {
        int c = tid + p * 256;
        int row = c >> 3, col = (c & 7) * 4;
        float4 f = *(const float4*)(A + (size_t)(bm + row) * C_ + k0 + col);
        st4(&As[row * 48 + col], f2bf(f.x), f2bf(f.y), f2bf(f.z), f2bf(f.w));
      }
    } else {
      const __bf16* A = (const __bf16*)Ap;
#pragma unroll
      for (int p = 0; p < 4; ++p) {
        int c = tid + p * 256;
        int row = c >> 3, col = (c & 7) * 4;
        *(uint2*)&As[row * 48 + col] =
            *(const uint2*)(A + (size_t)(bm + row) * C_ + k0 + col);
      }
    }
    // ---- stage W tile transposed: BsT[n][k] ----
#pragma unroll
    for (int p = 0; p < 4; ++p) {
      int c = tid + p * 256;
      int row = c >> 5, col = (c & 31) * 4;       // row = k, col = n
      float4 f = *(const float4*)(W + (size_t)(k0 + row) * C_ + bn + col);
      BsT[(col + 0) * 48 + row] = f2bf(f.x);
      BsT[(col + 1) * 48 + row] = f2bf(f.y);
      BsT[(col + 2) * 48 + row] = f2bf(f.z);
      BsT[(col + 3) * 48 + row] = f2bf(f.w);
    }
    __syncthreads();

    // ---- A fragments: 2 x ds_load_b128 each ----
    v16bf af[2];
#pragma unroll
    for (int mf = 0; mf < 2; ++mf) {
      const int r0 = (wm + mf * 16 + l16) * 48;
      af[mf] = cat8(ld8(&As[r0 + half * 8]), ld8(&As[r0 + 16 + half * 8]));
    }
    // ---- B fragments (from transposed tile) + WMMA ----
#pragma unroll
    for (int nf = 0; nf < 4; ++nf) {
      const int c0 = (wn + nf * 16 + l16) * 48;
      v16bf bfr = cat8(ld8(&BsT[c0 + half * 8]), ld8(&BsT[c0 + 16 + half * 8]));
#pragma unroll
      for (int mf = 0; mf < 2; ++mf)
        acc[mf][nf] = __builtin_amdgcn_wmma_f32_16x16x32_bf16(
            false, af[mf], false, bfr, (short)0, acc[mf][nf], false, false);
    }
  }

  // ---- epilogue ----
#pragma unroll
  for (int mf = 0; mf < 2; ++mf) {
#pragma unroll
    for (int nf = 0; nf < 4; ++nf) {
      const int gc = bn + wn + nf * 16 + l16;
      const float bv = bias[gc];
#pragma unroll
      for (int r = 0; r < 8; ++r) {
        const int gm = bm + wm + mf * 16 + half * 8 + r;
        const float v = acc[mf][nf][r] + bv;
        if constexpr (MODE == 0) {
          __bf16* out = (__bf16*)outp;
          const int b = gm >> 11, t = gm & (T_ - 1);
          const int h = gc >> 6, d = gc & (D_ - 1);
          out[(((size_t)(b * H_ + h)) * T_ + t) * D_ + d] = f2bf(v);
        } else {
          float* out = (float*)outp;
          out[(size_t)gm * C_ + gc] = v;
        }
      }
    }
  }
}

// ---------------------------------------------------------------------------
// In-place RoPE on bf16 [B,H,T,D]; one thread per (d, d+32) pair.
// ---------------------------------------------------------------------------
__global__ __launch_bounds__(256)
void rope_wmma(__bf16* __restrict__ X, int n)  // n = B*H*T*(D/2)
{
  int i = blockIdx.x * blockDim.x + threadIdx.x;
  if (i >= n) return;
  const int p   = i & 31;
  const int row = i >> 5;             // bh*T + t
  const int t   = row & (T_ - 1);
  const float inv = __powf(10000.0f, -((float)(2 * p)) * (1.0f / 64.0f));
  float sn, cs;
  __sincosf((float)t * inv, &sn, &cs);
  const size_t base = (size_t)row * 64 + p;
  const float x1 = bf2f(X[base]);
  const float x2 = bf2f(X[base + 32]);
  X[base]      = f2bf(x1 * cs - x2 * sn);
  X[base + 32] = f2bf(x2 * cs + x1 * sn);
}

// ---------------------------------------------------------------------------
// Flash attention, sliding window j >= i-511 (non-causal).
// 128 threads = 4 waves, 64-query block, K/V tiles of 64 in LDS.
// K tile: row-major (async DMA when available). V tile: transposed so the
// PV B-operand reads are ds_load_b128. P round-trips through LDS.
// ---------------------------------------------------------------------------
__global__ __launch_bounds__(128)
void attn_wmma(const __bf16* __restrict__ Qh, const __bf16* __restrict__ Kh,
               const __bf16* __restrict__ Vh, __bf16* __restrict__ Ah)
{
  __shared__ __attribute__((aligned(16))) __bf16 Kt [64 * 80];      // [key][d]
  __shared__ __attribute__((aligned(16))) __bf16 VtT[64 * 80];      // [d][key]
  __shared__ __attribute__((aligned(16))) __bf16 Pt [4 * 16 * 80];  // [m][key]

  const int q0 = blockIdx.x * 64;
  const int bh = blockIdx.y;                 // b*16 + h
  const __bf16* Qb = Qh + (size_t)bh * T_ * D_;
  const __bf16* Kb = Kh + (size_t)bh * T_ * D_;
  const __bf16* Vb = Vh + (size_t)bh * T_ * D_;

  const int tid = threadIdx.x, lane = tid & 31, w = tid >> 5;
  const int half = lane >> 4, l16 = lane & 15;
  const int qw = q0 + w * 16;                // this wave's 16 query rows

  // Q fragments (A operand): 2 x global_load_b128 each, loop-resident
  v16bf qf[2];
#pragma unroll
  for (int kc = 0; kc < 2; ++kc) {
    const __bf16* qp = Qb + (size_t)(qw + l16) * D_ + kc * 32;
    qf[kc] = cat8(ld8(qp + half * 8), ld8(qp + 16 + half * 8));
  }

  v8f o[4];
  for (int f = 0; f < 4; ++f) o[f] = zero8();
  float mrow[8], lrow[8];
#pragma unroll
  for (int r = 0; r < 8; ++r) { mrow[r] = NEGF; lrow[r] = 0.f; }

  int jmin = q0 - (WIN_ - 1); if (jmin < 0) jmin = 0;
  const int kt0 = (jmin >> 6) << 6;

  for (int kt = kt0; kt < T_; kt += 64) {
    __syncthreads();
#pragma unroll
    for (int p = 0; p < 4; ++p) {
      int c = tid + p * 128;
      int row = c >> 3, col = (c & 7) * 8;        // row = key, col = d
#if USE_ASYNC_COPY
      async_cp16(Kb + (size_t)(kt + row) * D_ + col, &Kt[row * 80 + col]);
#else
      *(uint4*)&Kt[row * 80 + col] =
          *(const uint4*)(Kb + (size_t)(kt + row) * D_ + col);
#endif
      // V transposed staging (DMA cannot transpose)
      union { uint4 q; __bf16 e[8]; } vv;
      vv.q = *(const uint4*)(Vb + (size_t)(kt + row) * D_ + col);
#pragma unroll
      for (int j = 0; j < 8; ++j) VtT[(col + j) * 80 + row] = vv.e[j];
    }
#if USE_ASYNC_COPY
    __builtin_amdgcn_s_wait_asynccnt(0);
#endif
    __syncthreads();

    // S = Q K^T * 1/sqrt(D)
    v8f s[4];
#pragma unroll
    for (int f = 0; f < 4; ++f) {
      v8f sa = zero8();
#pragma unroll
      for (int kc = 0; kc < 2; ++kc) {
        const int krow = (f * 16 + l16) * 80 + kc * 32;
        v16bf kfr = cat8(ld8(&Kt[krow + half * 8]),
                         ld8(&Kt[krow + 16 + half * 8]));
        sa = __builtin_amdgcn_wmma_f32_16x16x32_bf16(false, qf[kc], false, kfr,
                                                     (short)0, sa, false, false);
      }
#pragma unroll
      for (int r = 0; r < 8; ++r) s[f][r] = sa[r] * 0.125f;
    }

    // sliding-window mask only on boundary tiles
    if (kt < q0 - 448) {
#pragma unroll
      for (int f = 0; f < 4; ++f) {
        const int j = kt + f * 16 + l16;
#pragma unroll
        for (int r = 0; r < 8; ++r) {
          const int i = qw + half * 8 + r;
          if (j < i - (WIN_ - 1)) s[f][r] = NEGF;
        }
      }
    }

    // row max across the 16 N-lanes of each half-group
    float rmax[8];
#pragma unroll
    for (int r = 0; r < 8; ++r) {
      float m0 = fmaxf(fmaxf(s[0][r], s[1][r]), fmaxf(s[2][r], s[3][r]));
      m0 = fmaxf(m0, __shfl_xor(m0, 1));
      m0 = fmaxf(m0, __shfl_xor(m0, 2));
      m0 = fmaxf(m0, __shfl_xor(m0, 4));
      m0 = fmaxf(m0, __shfl_xor(m0, 8));
      rmax[r] = m0;
    }

    float al[8], rsum[8];
#pragma unroll
    for (int r = 0; r < 8; ++r) {
      const float mn = fmaxf(mrow[r], rmax[r]);
      al[r] = __expf(mrow[r] - mn);
      mrow[r] = mn;
      rsum[r] = 0.f;
    }

    // P = exp(S - m); route through LDS to re-layout C/D -> A operand
    const int pbase = w * 16 * 80;
#pragma unroll
    for (int f = 0; f < 4; ++f) {
#pragma unroll
      for (int r = 0; r < 8; ++r) {
        const float pv = __expf(s[f][r] - mrow[r]);
        rsum[r] += pv;
        Pt[pbase + (half * 8 + r) * 80 + f * 16 + l16] = f2bf(pv);
      }
    }
#pragma unroll
    for (int r = 0; r < 8; ++r) {
      float t0 = rsum[r];
      t0 += __shfl_xor(t0, 1);
      t0 += __shfl_xor(t0, 2);
      t0 += __shfl_xor(t0, 4);
      t0 += __shfl_xor(t0, 8);
      lrow[r] = lrow[r] * al[r] + t0;
    }
#pragma unroll
    for (int f = 0; f < 4; ++f)
#pragma unroll
      for (int r = 0; r < 8; ++r) o[f][r] *= al[r];

    // O += P V   (same-wave LDS RAW: DS ops are in-order per wave)
#pragma unroll
    for (int kc = 0; kc < 2; ++kc) {
      const int prow = pbase + l16 * 80 + kc * 32;
      v16bf pf = cat8(ld8(&Pt[prow + half * 8]), ld8(&Pt[prow + 16 + half * 8]));
#pragma unroll
      for (int f = 0; f < 4; ++f) {
        const int vrow = (f * 16 + l16) * 80 + kc * 32;
        v16bf vfr = cat8(ld8(&VtT[vrow + half * 8]),
                         ld8(&VtT[vrow + 16 + half * 8]));
        o[f] = __builtin_amdgcn_wmma_f32_16x16x32_bf16(false, pf, false, vfr,
                                                       (short)0, o[f], false, false);
      }
    }
  }

  // normalize and write attention output as bf16 in [B,T,C] for the O-proj
  const int b = bh >> 4, h = bh & 15;
#pragma unroll
  for (int f = 0; f < 4; ++f) {
#pragma unroll
    for (int r = 0; r < 8; ++r) {
      const int t = qw + half * 8 + r;
      const int d = f * 16 + l16;
      const float v = o[f][r] / lrow[r];
      Ah[((size_t)(b * T_ + t)) * C_ + h * 64 + d] = f2bf(v);
    }
  }
}

// ---------------------------------------------------------------------------
extern "C" void kernel_launch(void* const* d_in, const int* in_sizes, int n_in,
                              void* d_out, int out_size, void* d_ws, size_t ws_size,
                              hipStream_t stream)
{
  (void)in_sizes; (void)n_in; (void)out_size; (void)ws_size;

  const float* x  = (const float*)d_in[0];
  const float* Wq = (const float*)d_in[1];
  const float* bq = (const float*)d_in[2];
  const float* Wk = (const float*)d_in[3];
  const float* bk = (const float*)d_in[4];
  const float* Wv = (const float*)d_in[5];
  const float* bv = (const float*)d_in[6];
  const float* Wo = (const float*)d_in[7];
  const float* bo = (const float*)d_in[8];

  // workspace: Qh, Kh, Vh (head-major bf16), Ah (row-major bf16) = 32 MB
  char* ws = (char*)d_ws;
  const size_t nBHTD = (size_t)B_ * H_ * T_ * D_;   // 4,194,304
  __bf16* Qh = (__bf16*)ws; ws += nBHTD * 2;
  __bf16* Kh = (__bf16*)ws; ws += nBHTD * 2;
  __bf16* Vh = (__bf16*)ws; ws += nBHTD * 2;
  __bf16* Ah = (__bf16*)ws; ws += nBHTD * 2;

  const dim3 gGrid(C_ / 128, M_ / 128);   // (8, 32)

  gemm_wmma<0><<<gGrid, 256, 0, stream>>>((const void*)x, Wq, bq, (void*)Qh);
  gemm_wmma<0><<<gGrid, 256, 0, stream>>>((const void*)x, Wk, bk, (void*)Kh);
  gemm_wmma<0><<<gGrid, 256, 0, stream>>>((const void*)x, Wv, bv, (void*)Vh);

  const int nrope = B_ * H_ * T_ * (D_ / 2);   // 2,097,152
  rope_wmma<<<nrope / 256, 256, 0, stream>>>(Qh, nrope);
  rope_wmma<<<nrope / 256, 256, 0, stream>>>(Kh, nrope);

  attn_wmma<<<dim3(T_ / 64, B_ * H_), 128, 0, stream>>>(Qh, Kh, Vh, Ah);

  gemm_wmma<1><<<gGrid, 256, 0, stream>>>((const void*)Ah, Wo, bo, d_out);
}
